// Conv2dAct_51573967290882
// MI455X (gfx1250) — matrix operand order repaired and checked by reading, not compiled, not run on hardware
//
#include <hip/hip_runtime.h>

typedef _Float16 v16h __attribute__((ext_vector_type(16)));
typedef float    v8f  __attribute__((ext_vector_type(8)));

#define N_IMG   16
#define C_IN    256
#define H_IN    128
#define C_OUT   512
#define H_BLUR  129
#define H_OUT   64
#define KTOT    2304      // 9 taps * 256 channels
#define KSTEPS  (KTOT / 32)
#define LDSB_STRIDE 40    // 32 f16 payload + 8 f16 pad (80B rows: conflict-free)

static constexpr size_t ACT_ELEMS = (size_t)N_IMG * H_BLUR * H_BLUR * C_IN;
static constexpr size_t ACT_BYTES = ACT_ELEMS * sizeof(_Float16);

// ---------------------------------------------------------------- blur (FIR)
// x: NCHW f32 [16,256,128,128]  ->  act: NHWC f16 [16,129,129,256]
__global__ __launch_bounds__(256) void blur_nhwc_f16(const float* __restrict__ xin,
                                                     const float* __restrict__ fil,
                                                     _Float16* __restrict__ act) {
  const int c = threadIdx.x;          // 0..255 (channel, contiguous in output)
  int sp = blockIdx.x;                // n*129*129 + y*129 + x
  const int x = sp % H_BLUR; sp /= H_BLUR;
  const int y = sp % H_BLUR;
  const int n = sp / H_BLUR;

  const float* base = xin + ((size_t)(n * C_IN + c)) * (H_IN * H_IN);
  float acc = 0.f;
#pragma unroll
  for (int i = 0; i < 4; ++i) {
    const int iy = y + i - 2;
    if (iy < 0 || iy >= H_IN) continue;
#pragma unroll
    for (int j = 0; j < 4; ++j) {
      const int ix = x + j - 2;
      if (ix < 0 || ix >= H_IN) continue;
      acc += base[iy * H_IN + ix] * fil[i * 4 + j];
    }
  }
  act[(size_t)blockIdx.x * C_IN + c] = (_Float16)acc;
}

// ------------------------------------------------------------- weight prep
// w: [512,256,3,3] f32 -> wT: [oc][tap*256+c] f16, pre-scaled by 1/48
__global__ __launch_bounds__(256) void prep_w(const float* __restrict__ w,
                                              _Float16* __restrict__ wT) {
  const int idx = blockIdx.x * 256 + threadIdx.x;
  if (idx >= C_OUT * KTOT) return;
  const int oc = idx / KTOT;
  const int k  = idx % KTOT;
  const int tap = k >> 8;             // 0..8
  const int c   = k & 255;
  const int ky = tap / 3, kx = tap % 3;
  wT[idx] = (_Float16)(w[((oc * C_IN + c) * 3 + ky) * 3 + kx] * 0.020833333333333332f);
}

// ----------------------------------------------------------- CDNA5 helpers
__device__ __forceinline__ void async_b128(_Float16* lds_dst, const _Float16* gsrc) {
  // Async DMA path: per-lane 16B copy global -> LDS, tracked by ASYNCcnt.
  const unsigned loff = (unsigned)(unsigned long long)(uintptr_t)lds_dst; // LDS offset
  asm volatile("global_load_async_to_lds_b128 %0, %1, off"
               :: "v"(loff), "v"(gsrc) : "memory");
}
__device__ __forceinline__ void wait_async0() {
  asm volatile("s_wait_asynccnt 0" ::: "memory");
}

// A-fragment (16x32 f16): one row, 16 contiguous K values K[h*16..h*16+15]
__device__ __forceinline__ v16h ld_a_frag(const _Float16* p) {
  v16h r;
  float4* rp = reinterpret_cast<float4*>(&r);
  rp[0] = *reinterpret_cast<const float4*>(p);        // K chunk 0
  rp[1] = *reinterpret_cast<const float4*>(p + 8);    // K chunk +8
  return r;
}
// B-fragment (32x16 f16): one column, 16 contiguous K values
__device__ __forceinline__ v16h ld_b_frag(const _Float16* p) {
  v16h r;
  float4* rp = reinterpret_cast<float4*>(&r);
  rp[0] = *reinterpret_cast<const float4*>(p);
  rp[1] = *reinterpret_cast<const float4*>(p + 8);
  return r;
}

// ------------------------------------------------------ implicit-GEMM WMMA
// D[oc][pixel] = sum_k wT[oc][k] * act_im2col[k][pixel], K = (tap, channel)
// Block: 64 oc x 128 pixels; 8 waves as 2(oc) x 4(pix); wave = 32x32 = 2x2 WMMA.
// Double-buffered async staging of the activation tile (copy of tile i+1
// overlaps the 4 WMMAs + weight loads of tile i).
__global__ __launch_bounds__(256, 1) void conv_gemm_wmma(
    const _Float16* __restrict__ act, const _Float16* __restrict__ wT,
    const float* __restrict__ bias, float* __restrict__ out) {
  __shared__ _Float16 ldsB[2][128 * LDSB_STRIDE];   // 2 x 10 KB activation tiles

  const int tid  = threadIdx.x;
  const int lane = tid & 31;
  const int ll   = lane & 15;     // column/row within 16
  const int lh   = lane >> 4;     // lane half selects K/M sub-range
  const int wave = tid >> 5;
  const int wm   = wave & 1;      // oc subtile (x32)
  const int wn   = wave >> 1;     // pixel subtile (x32)

  const int r0  = blockIdx.x * 128;   // pixel tile base (covers 2 output rows)
  const int ocb = blockIdx.y * 64;    // out-channel tile base

  // fixed per-thread B-staging coords: thread copies 32B of one pixel row
  const int cp     = tid >> 1;        // pixel within tile 0..127
  const int cchunk = tid & 1;         // which 16-channel half
  const int gp  = r0 + cp;
  const int cn  = gp >> 12;           // image
  const int coy = (gp >> 6) & 63;     // output row
  const int cox = gp & 63;            // output col
  // source base for this thread's pixel at tap (0,0), channel chunk
  const _Float16* src_base =
      act + (((size_t)cn * H_BLUR + 2 * coy) * H_BLUR + 2 * cox) * C_IN + cchunk * 16;

  const _Float16* wrow0 = wT + (size_t)(ocb + wm * 32 + ll) * KTOT;
  const _Float16* wrow1 = wrow0 + (size_t)16 * KTOT;

  v8f acc00 = {}, acc01 = {}, acc10 = {}, acc11 = {};

  // issue async staging of K-block kk into buffer buf
  auto issue_tile = [&](int kk, int buf) {
    const int tap = kk >> 8;          // spatial tap 0..8
    const int c0  = kk & 255;         // channel block
    const int ky  = tap / 3;
    const int kx  = tap % 3;
    const _Float16* src = src_base + ((size_t)ky * H_BLUR + kx) * C_IN + c0;
    _Float16* dst = &ldsB[buf][cp * LDSB_STRIDE + cchunk * 16];
    async_b128(dst,     src);
    async_b128(dst + 8, src + 8);
  };

  issue_tile(0, 0);                   // prologue prefetch

  for (int i = 0; i < KSTEPS; ++i) {
    const int kk  = i * 32;
    const int cur = i & 1;

    wait_async0();                    // my tile-i copies done (issued 1 iter ago)
    __syncthreads();                  // everyone's tile-i in LDS; buf[cur^1] reads retired

    if (i + 1 < KSTEPS) issue_tile(kk + 32, cur ^ 1);   // overlaps with compute below

    // A fragments: weights direct from L2-resident global
    const v16h a0 = ld_a_frag(wrow0 + kk + lh * 16);
    const v16h a1 = ld_a_frag(wrow1 + kk + lh * 16);
    // B fragments: activation columns from LDS
    const v16h b0 = ld_b_frag(&ldsB[cur][(wn * 32      + ll) * LDSB_STRIDE + lh * 16]);
    const v16h b1 = ld_b_frag(&ldsB[cur][(wn * 32 + 16 + ll) * LDSB_STRIDE + lh * 16]);

    acc00 = __builtin_amdgcn_wmma_f32_16x16x32_f16(false, a0, false, b0, (short)0, acc00, false, false);
    acc01 = __builtin_amdgcn_wmma_f32_16x16x32_f16(false, a0, false, b1, (short)0, acc01, false, false);
    acc10 = __builtin_amdgcn_wmma_f32_16x16x32_f16(false, a1, false, b0, (short)0, acc10, false, false);
    acc11 = __builtin_amdgcn_wmma_f32_16x16x32_f16(false, a1, false, b1, (short)0, acc11, false, false);
  }

  // epilogue: bias + lrelu(0.2) * sqrt(2); D layout: M = vgpr + 8*lh, N = ll
  const float gain = 1.41421356237f;
#pragma unroll
  for (int sm = 0; sm < 2; ++sm) {
#pragma unroll
    for (int sn = 0; sn < 2; ++sn) {
      const v8f a = (sm == 0) ? (sn == 0 ? acc00 : acc01)
                              : (sn == 0 ? acc10 : acc11);
      const int pix = r0 + wn * 32 + sn * 16 + ll;   // consecutive ox across lanes
      const int n  = pix >> 12;
      const int oy = (pix >> 6) & 63;
      const int ox = pix & 63;
#pragma unroll
      for (int j = 0; j < 8; ++j) {
        const int oc = ocb + wm * 32 + sm * 16 + lh * 8 + j;
        float v = a[j] + bias[oc];
        v = (v >= 0.f ? v : 0.2f * v) * gain;
        out[(((size_t)n * C_OUT + oc) * H_OUT + oy) * H_OUT + ox] = v;
      }
    }
  }
}

// ------------------------------------------------------------------ launch
extern "C" void kernel_launch(void* const* d_in, const int* in_sizes, int n_in,
                              void* d_out, int out_size, void* d_ws, size_t ws_size,
                              hipStream_t stream) {
  (void)in_sizes; (void)n_in; (void)out_size; (void)ws_size;
  const float* x    = (const float*)d_in[0];
  const float* w    = (const float*)d_in[1];
  const float* bias = (const float*)d_in[2];
  const float* fil  = (const float*)d_in[3];

  _Float16* act = (_Float16*)d_ws;                       // 136.3 MB NHWC f16
  _Float16* wT  = (_Float16*)((char*)d_ws + ACT_BYTES);  // 2.36 MB f16
  float* out = (float*)d_out;

  blur_nhwc_f16<<<N_IMG * H_BLUR * H_BLUR, 256, 0, stream>>>(x, fil, act);
  prep_w<<<(C_OUT * KTOT + 255) / 256, 256, 0, stream>>>(w, wT);
  conv_gemm_wmma<<<dim3(512, 8), 256, 0, stream>>>(act, wT, bias, out);
}
